// HierarchicalEEGTransformer_71751723647689
// MI455X (gfx1250) — matrix operand order repaired and checked by reading, not compiled, not run on hardware
//
#include <hip/hip_runtime.h>
#include <hip/hip_bf16.h>

typedef __attribute__((ext_vector_type(8)))  __bf16 v8bf;
typedef __attribute__((ext_vector_type(16))) __bf16 v16bf;
typedef __attribute__((ext_vector_type(8)))  float  v8f;
typedef __attribute__((ext_vector_type(4)))  int    v4i;

#define NLAYER 6
#define HEADS  12
#define WIN    2304
#define DMODEL 768
#define DFF    3072
#define VOCAB  4096
#define SEQ    4608
#define HD     64

#ifndef __has_builtin
#define __has_builtin(x) 0
#endif
#if __has_builtin(__builtin_amdgcn_global_load_async_to_lds_b128)
#define USE_ASYNC_LDS 1
#else
#define USE_ASYNC_LDS 0
#endif

#define AS1 __attribute__((address_space(1)))
#define AS3 __attribute__((address_space(3)))

__device__ __forceinline__ void wait_async0() {
#if USE_ASYNC_LDS
#if __has_builtin(__builtin_amdgcn_s_wait_asynccnt)
  __builtin_amdgcn_s_wait_asynccnt(0);
#else
  asm volatile("s_wait_asynccnt 0x0" ::: "memory");
#endif
#endif
}

__device__ __forceinline__ __bf16 f2bf(float f) { return (__bf16)f; }

__device__ __forceinline__ v16bf cat16(v8bf lo, v8bf hi) {
  return __builtin_shufflevector(lo, hi, 0,1,2,3,4,5,6,7,8,9,10,11,12,13,14,15);
}

__device__ __forceinline__ v8f zero8() {
  v8f z;
#pragma unroll
  for (int i = 0; i < 8; ++i) z[i] = 0.0f;
  return z;
}

__device__ __forceinline__ v8f wmma_bf16(v16bf a, v16bf b, v8f c) {
  return __builtin_amdgcn_wmma_f32_16x16x32_bf16(false, a, false, b, (short)0, c,
                                                 false, false);
}

// xor-mask lane swap within groups of 16 via ds_swizzle (imm-encoded)
template <int XM>
__device__ __forceinline__ float swz_xor(float v) {
  return __int_as_float(
      __builtin_amdgcn_ds_swizzle(__float_as_int(v), (XM << 10) | 0x1f));
}
__device__ __forceinline__ float reduce_max16(float v) {
  v = fmaxf(v, swz_xor<1>(v));
  v = fmaxf(v, swz_xor<2>(v));
  v = fmaxf(v, swz_xor<4>(v));
  v = fmaxf(v, swz_xor<8>(v));
  return v;
}
__device__ __forceinline__ float reduce_sum16(float v) {
  v += swz_xor<1>(v);
  v += swz_xor<2>(v);
  v += swz_xor<4>(v);
  v += swz_xor<8>(v);
  return v;
}

// ---------------------------------------------------------------------------
// Transpose+convert weights: src f32 [L][K][N] -> dst bf16 [L][N][K]
// ---------------------------------------------------------------------------
__global__ __launch_bounds__(256) void convt_kernel(
    const float* __restrict__ src, __bf16* __restrict__ dst,
    int Kd, int Nd, int total) {
  int i = blockIdx.x * 256 + threadIdx.x;
  if (i >= total) return;
  int per = Kd * Nd;
  int l = i / per, r = i - l * per;
  int n = r / Kd, k = r - n * Kd;
  dst[i] = f2bf(src[(size_t)l * per + (size_t)k * Nd + n]);
}

// ---------------------------------------------------------------------------
// Embedding: h = tok_emb[x] + win_pos[row/W] + tok_pos[row%W]   (f32)
// ---------------------------------------------------------------------------
__global__ __launch_bounds__(256) void embed_kernel(
    const int* __restrict__ x, const float* __restrict__ tok_emb,
    const float* __restrict__ win_pos, const float* __restrict__ tok_pos,
    float* __restrict__ h) {
  int i = blockIdx.x * 256 + threadIdx.x;
  if (i >= SEQ * DMODEL) return;
  int row = i / DMODEL, d = i - row * DMODEL;
  int tok = x[row];
  h[i] = tok_emb[(size_t)tok * DMODEL + d] +
         win_pos[(size_t)(row / WIN) * DMODEL + d] +
         tok_pos[(size_t)(row % WIN) * DMODEL + d];
}

// ---------------------------------------------------------------------------
// LayerNorm over D=768, one 256-thread block per row, writes bf16
// ---------------------------------------------------------------------------
__global__ __launch_bounds__(256) void ln_kernel(
    const float* __restrict__ x, const float* __restrict__ g,
    const float* __restrict__ b, __bf16* __restrict__ y) {
  __shared__ float red[256];
  const int row = blockIdx.x, tid = threadIdx.x;
  const float* xr = x + (size_t)row * DMODEL;
  float s = 0.0f;
  for (int i = tid; i < DMODEL; i += 256) s += xr[i];
  red[tid] = s; __syncthreads();
  for (int off = 128; off > 0; off >>= 1) {
    if (tid < off) red[tid] += red[tid + off];
    __syncthreads();
  }
  float mean = red[0] * (1.0f / DMODEL);
  __syncthreads();
  float vs = 0.0f;
  for (int i = tid; i < DMODEL; i += 256) { float d = xr[i] - mean; vs += d * d; }
  red[tid] = vs; __syncthreads();
  for (int off = 128; off > 0; off >>= 1) {
    if (tid < off) red[tid] += red[tid + off];
    __syncthreads();
  }
  float rstd = rsqrtf(red[0] * (1.0f / DMODEL) + 1e-5f);
  __bf16* yr = y + (size_t)row * DMODEL;
  for (int i = tid; i < DMODEL; i += 256)
    yr[i] = f2bf((xr[i] - mean) * rstd * g[i] + b[i]);
}

// ---------------------------------------------------------------------------
// V transpose: vt[f][s] = v[s][f]   (bf16)
// ---------------------------------------------------------------------------
__global__ __launch_bounds__(256) void vtrans_kernel(
    const __bf16* __restrict__ v, __bf16* __restrict__ vt) {
  int i = blockIdx.x * 256 + threadIdx.x;
  if (i >= SEQ * DMODEL) return;
  int f = i / SEQ, s = i - f * SEQ;
  vt[i] = v[(size_t)s * DMODEL + f];
}

// ---------------------------------------------------------------------------
// Tiled WMMA GEMM: C[M,N] = A[M,K](bf16,row-major) * Bt[N,K](bf16) + bias
// Block 128x128, k-slab 64, double-buffered LDS, 8 waves (2x4), each wave
// 4x2 16x16 C-tiles (16 WMMA per slab). Async global->LDS copies when the
// toolchain exposes them (ASYNCcnt path), register-staged copies otherwise.
// ---------------------------------------------------------------------------
template <bool GELU, bool RES, bool OUTBF>
__global__ __launch_bounds__(256) void gemm_kernel(
    const __bf16* __restrict__ A, const __bf16* __restrict__ Bt,
    const float* __restrict__ bias, const float* res,
    float* outf, __bf16* outb, int M, int N, int K) {
  __shared__ __align__(64) __bf16 lds_a[2][128 * 64];   // [m][k]
  __shared__ __align__(64) __bf16 lds_b[2][128 * 64];   // [n][k]
  const int tid  = threadIdx.x;
  const int lane = tid & 31;
  const int w    = tid >> 5;
  const int half = lane >> 4;
  const int lm   = lane & 15;
  const int wm   = w >> 2;        // 0..1
  const int wn   = w & 3;         // 0..3
  const int m_blk = blockIdx.y * 128;
  const int n_blk = blockIdx.x * 128;
  const int KT = K >> 6;          // K/64 slabs

  v8f acc[4][2];
#pragma unroll
  for (int mt = 0; mt < 4; ++mt)
#pragma unroll
    for (int nt = 0; nt < 2; ++nt) acc[mt][nt] = zero8();

#if !USE_ASYNC_LDS
  v8bf ra[4], rb[4];
#endif

  // stage slab kt into buffer buf: 128 rows x 64 bf16 each for A and B
  auto issue_stage = [&](int kt, int buf) {
    const __bf16* Ab = A  + (size_t)m_blk * K + kt * 64;
    const __bf16* Bb = Bt + (size_t)n_blk * K + kt * 64;
#pragma unroll
    for (int i = 0; i < 4; ++i) {
      int c = tid + 256 * i;          // 0..1023
      int row = c >> 3, cc = c & 7;   // 8 chunks of 8 bf16 per 64-wide row
#if USE_ASYNC_LDS
      __builtin_amdgcn_global_load_async_to_lds_b128(
          (AS1 v4i*)(v4i*)(Ab + (size_t)row * K + cc * 8),
          (AS3 v4i*)(v4i*)&lds_a[buf][row * 64 + cc * 8], 0, 0);
      __builtin_amdgcn_global_load_async_to_lds_b128(
          (AS1 v4i*)(v4i*)(Bb + (size_t)row * K + cc * 8),
          (AS3 v4i*)(v4i*)&lds_b[buf][row * 64 + cc * 8], 0, 0);
#else
      ra[i] = *(const v8bf*)(Ab + (size_t)row * K + cc * 8);
      rb[i] = *(const v8bf*)(Bb + (size_t)row * K + cc * 8);
#endif
    }
  };
  auto commit_stage = [&](int buf) {
#if USE_ASYNC_LDS
    (void)buf;
    wait_async0();
#else
#pragma unroll
    for (int i = 0; i < 4; ++i) {
      int c = tid + 256 * i;
      int row = c >> 3, cc = c & 7;
      *(v8bf*)(&lds_a[buf][row * 64 + cc * 8]) = ra[i];
      *(v8bf*)(&lds_b[buf][row * 64 + cc * 8]) = rb[i];
    }
#endif
  };
  auto compute = [&](int buf) {
#pragma unroll
    for (int kc = 0; kc < 2; ++kc) {
      v16bf af[4], bfr[2];
#pragma unroll
      for (int mt = 0; mt < 4; ++mt) {
        const __bf16* p =
            &lds_a[buf][(wm * 64 + mt * 16 + lm) * 64 + kc * 32 + 8 * half];
        af[mt] = cat16(*(const v8bf*)p, *(const v8bf*)(p + 16));
      }
#pragma unroll
      for (int nt = 0; nt < 2; ++nt) {
        const __bf16* p =
            &lds_b[buf][(wn * 32 + nt * 16 + lm) * 64 + kc * 32 + 8 * half];
        bfr[nt] = cat16(*(const v8bf*)p, *(const v8bf*)(p + 16));
      }
#pragma unroll
      for (int mt = 0; mt < 4; ++mt)
#pragma unroll
        for (int nt = 0; nt < 2; ++nt)
          acc[mt][nt] = wmma_bf16(af[mt], bfr[nt], acc[mt][nt]);
    }
  };

  // software pipeline: stage(i+1) overlaps compute(i); 1 barrier per slab
  issue_stage(0, 0);
  commit_stage(0);
  __syncthreads();
  for (int kt = 0; kt < KT; ++kt) {
    int cur = kt & 1, nxt = cur ^ 1;
    if (kt + 1 < KT) issue_stage(kt + 1, nxt);
    compute(cur);
    if (kt + 1 < KT) commit_stage(nxt);
    __syncthreads();
  }

  // Epilogue: bias (+GELU) (+residual), f32 or bf16 store
#pragma unroll
  for (int mt = 0; mt < 4; ++mt) {
#pragma unroll
    for (int nt = 0; nt < 2; ++nt) {
      int gn = n_blk + wn * 32 + nt * 16 + lm;
      float bv = bias[gn];
#pragma unroll
      for (int r = 0; r < 8; ++r) {
        int gm = m_blk + wm * 64 + mt * 16 + r + 8 * half;
        float v = acc[mt][nt][r] + bv;
        if constexpr (GELU) v = 0.5f * v * (1.0f + erff(v * 0.70710678118f));
        if constexpr (RES)  v += res[(size_t)gm * N + gn];
        if constexpr (OUTBF) outb[(size_t)gm * N + gn] = f2bf(v);
        else                 outf[(size_t)gm * N + gn] = v;
      }
    }
  }
}

// ---------------------------------------------------------------------------
// Flash attention: one wave per (head, 16-row Q tile), hd=64, 64 keys/iter.
// Block mask == kv range [0, (qwin+1)*W). All GEMMs via WMMA bf16.
// Q,K: [S][D] bf16 (head h at cols h*64..). VT: [D][S] bf16.
// ---------------------------------------------------------------------------
__global__ __launch_bounds__(256) void attn_kernel(
    const __bf16* __restrict__ Q, const __bf16* __restrict__ K,
    const __bf16* __restrict__ VT, __bf16* __restrict__ CTX) {
  __shared__ __align__(64) __bf16 pbuf[8][16 * 64];  // per-wave P staging
  const int lane = threadIdx.x & 31;
  const int w    = threadIdx.x >> 5;
  const int half = lane >> 4;
  const int lm   = lane & 15;
  const int qtiles = SEQ / 16;                       // 288
  const int gw    = blockIdx.x * 8 + w;
  const int head  = gw / qtiles;
  const int qtile = gw % qtiles;
  const int qbase = qtile * 16;
  const int kv_end = ((qbase / WIN) + 1) * WIN;      // multiple of 64
  const float scale = 0.125f;                        // 1/sqrt(64)

  const __bf16* qrow = Q + (size_t)(qbase + lm) * DMODEL + head * HD;
  v16bf qa0 = cat16(*(const v8bf*)(qrow +  0 + 16 * half),
                    *(const v8bf*)(qrow +  8 + 16 * half));
  v16bf qa1 = cat16(*(const v8bf*)(qrow + 32 + 16 * half),
                    *(const v8bf*)(qrow + 40 + 16 * half));

  float mrow[8], lrow[8];
  v8f ctx[4];
#pragma unroll
  for (int r = 0; r < 8; ++r) { mrow[r] = -1e30f; lrow[r] = 0.0f; }
#pragma unroll
  for (int t = 0; t < 4; ++t) ctx[t] = zero8();

  for (int kb = 0; kb < kv_end; kb += 64) {
    // 4 score tiles: keys [kb+16t, kb+16t+16)
    v8f s[4];
#pragma unroll
    for (int t = 0; t < 4; ++t) {
      const __bf16* kr =
          K + (size_t)(kb + t * 16 + lm) * DMODEL + head * HD + 16 * half;
      s[t] = zero8();
      s[t] = wmma_bf16(qa0, *(const v16bf*)(kr), s[t]);
      s[t] = wmma_bf16(qa1, *(const v16bf*)(kr + 32), s[t]);
    }
    // online softmax over 64 keys at once
#pragma unroll
    for (int r = 0; r < 8; ++r) {
      float v0 = s[0][r] * scale, v1 = s[1][r] * scale;
      float v2 = s[2][r] * scale, v3 = s[3][r] * scale;
      float mx = reduce_max16(fmaxf(fmaxf(v0, v1), fmaxf(v2, v3)));
      float newm = fmaxf(mrow[r], mx);
      float alpha = __expf(mrow[r] - newm);
      float p0 = __expf(v0 - newm), p1 = __expf(v1 - newm);
      float p2 = __expf(v2 - newm), p3 = __expf(v3 - newm);
      float rs = reduce_sum16((p0 + p1) + (p2 + p3));
      lrow[r] = lrow[r] * alpha + rs;
      mrow[r] = newm;
#pragma unroll
      for (int t = 0; t < 4; ++t) ctx[t][r] *= alpha;
      int prow = r + 8 * half;
      pbuf[w][prow * 64 + lm]      = f2bf(p0);
      pbuf[w][prow * 64 + 16 + lm] = f2bf(p1);
      pbuf[w][prow * 64 + 32 + lm] = f2bf(p2);
      pbuf[w][prow * 64 + 48 + lm] = f2bf(p3);
    }
    // reload P as two A-fragments (DS ops from same wave stay in order)
    const __bf16* pp0 = &pbuf[w][lm * 64 + 16 * half];
    const __bf16* pp1 = &pbuf[w][lm * 64 + 32 + 16 * half];
    v16bf pa0 = cat16(*(const v8bf*)pp0, *(const v8bf*)(pp0 + 8));
    v16bf pa1 = cat16(*(const v8bf*)pp1, *(const v8bf*)(pp1 + 8));

    // ctx += P(16x64) * V(64x64); VT gives contiguous B-fragments
#pragma unroll
    for (int t = 0; t < 4; ++t) {
      const __bf16* vr =
          VT + (size_t)(head * HD + t * 16 + lm) * SEQ + kb + 16 * half;
      ctx[t] = wmma_bf16(pa0, *(const v16bf*)(vr), ctx[t]);
      ctx[t] = wmma_bf16(pa1, *(const v16bf*)(vr + 32), ctx[t]);
    }
  }

  // normalize & write bf16 ctx [S][D]
#pragma unroll
  for (int t = 0; t < 4; ++t) {
#pragma unroll
    for (int r = 0; r < 8; ++r) {
      int gm = qbase + r + 8 * half;
      int gn = head * HD + t * 16 + lm;
      CTX[(size_t)gm * DMODEL + gn] = f2bf(ctx[t][r] / lrow[r]);
    }
  }
}

// ---------------------------------------------------------------------------
extern "C" void kernel_launch(void* const* d_in, const int* in_sizes, int n_in,
                              void* d_out, int out_size, void* d_ws, size_t ws_size,
                              hipStream_t stream) {
  const int*   x       = (const int*)  d_in[0];
  const float* tok_emb = (const float*)d_in[1];
  const float* win_pos = (const float*)d_in[2];
  const float* tok_pos = (const float*)d_in[3];
  const float* Wq  = (const float*)d_in[4];
  const float* bq  = (const float*)d_in[5];
  const float* Wk  = (const float*)d_in[6];
  const float* bk  = (const float*)d_in[7];
  const float* Wv  = (const float*)d_in[8];
  const float* bv  = (const float*)d_in[9];
  const float* Wo  = (const float*)d_in[10];
  const float* bo  = (const float*)d_in[11];
  const float* ln1g = (const float*)d_in[12];
  const float* ln1b = (const float*)d_in[13];
  const float* W1  = (const float*)d_in[14];
  const float* b1  = (const float*)d_in[15];
  const float* W2  = (const float*)d_in[16];
  const float* b2  = (const float*)d_in[17];
  const float* ln2g = (const float*)d_in[18];
  const float* ln2b = (const float*)d_in[19];
  const float* lnfg = (const float*)d_in[20];
  const float* lnfb = (const float*)d_in[21];
  const float* Whead = (const float*)d_in[22];
  const float* bhead = (const float*)d_in[23];

  char* ws = (char*)d_ws;
  size_t off = 0;
  auto take = [&](size_t bytes) -> char* {
    char* p = ws + off;
    off += (bytes + 255) & ~(size_t)255;
    return p;
  };
  const size_t DD  = (size_t)DMODEL * DMODEL;
  const size_t DF  = (size_t)DMODEL * DFF;
  const size_t SD  = (size_t)SEQ * DMODEL;

  __bf16* wq_bf = (__bf16*)take(NLAYER * DD * 2);   // [N][K] transposed
  __bf16* wk_bf = (__bf16*)take(NLAYER * DD * 2);
  __bf16* wv_bf = (__bf16*)take(NLAYER * DD * 2);
  __bf16* wo_bf = (__bf16*)take(NLAYER * DD * 2);
  __bf16* w1_bf = (__bf16*)take(NLAYER * DF * 2);
  __bf16* w2_bf = (__bf16*)take(NLAYER * DF * 2);
  __bf16* wh_bf = (__bf16*)take((size_t)DMODEL * VOCAB * 2);
  float*  h     = (float*) take(SD * 4);
  __bf16* hn    = (__bf16*)take(SD * 2);
  __bf16* qb    = (__bf16*)take(SD * 2);
  __bf16* kb    = (__bf16*)take(SD * 2);
  __bf16* vb    = (__bf16*)take(SD * 2);
  __bf16* vt    = (__bf16*)take(SD * 2);
  __bf16* ctxb  = (__bf16*)take(SD * 2);
  __bf16* ffb   = (__bf16*)take((size_t)SEQ * DFF * 2);
  (void)ws_size; (void)in_sizes; (void)n_in; (void)out_size;

  // --- transpose+convert weights to bf16 [N][K] (one-time per launch) ---
  {
    int n = NLAYER * (int)DD, g = (n + 255) / 256;
    convt_kernel<<<g, 256, 0, stream>>>(Wq, wq_bf, DMODEL, DMODEL, n);
    convt_kernel<<<g, 256, 0, stream>>>(Wk, wk_bf, DMODEL, DMODEL, n);
    convt_kernel<<<g, 256, 0, stream>>>(Wv, wv_bf, DMODEL, DMODEL, n);
    convt_kernel<<<g, 256, 0, stream>>>(Wo, wo_bf, DMODEL, DMODEL, n);
    int n1 = NLAYER * (int)DF, g1 = (n1 + 255) / 256;
    convt_kernel<<<g1, 256, 0, stream>>>(W1, w1_bf, DMODEL, DFF, n1);
    convt_kernel<<<g1, 256, 0, stream>>>(W2, w2_bf, DFF, DMODEL, n1);
    int nh = DMODEL * VOCAB, gh = (nh + 255) / 256;
    convt_kernel<<<gh, 256, 0, stream>>>(Whead, wh_bf, DMODEL, VOCAB, nh);
  }

  // --- embeddings ---
  embed_kernel<<<(SEQ * DMODEL + 255) / 256, 256, 0, stream>>>(
      x, tok_emb, win_pos, tok_pos, h);

  const dim3 gD(DMODEL / 128, SEQ / 128);    // (6, 36)
  const dim3 gF(DFF / 128, SEQ / 128);       // (24, 36)
  const dim3 gV(VOCAB / 128, SEQ / 128);     // (32, 36)
  const int attn_blocks = (HEADS * (SEQ / 16)) / 8;  // 432

  for (int l = 0; l < NLAYER; ++l) {
    ln_kernel<<<SEQ, 256, 0, stream>>>(h, ln1g + l * DMODEL, ln1b + l * DMODEL, hn);
    gemm_kernel<false, false, true><<<gD, 256, 0, stream>>>(
        hn, wq_bf + l * DD, bq + l * DMODEL, nullptr, nullptr, qb,
        SEQ, DMODEL, DMODEL);
    gemm_kernel<false, false, true><<<gD, 256, 0, stream>>>(
        hn, wk_bf + l * DD, bk + l * DMODEL, nullptr, nullptr, kb,
        SEQ, DMODEL, DMODEL);
    gemm_kernel<false, false, true><<<gD, 256, 0, stream>>>(
        hn, wv_bf + l * DD, bv + l * DMODEL, nullptr, nullptr, vb,
        SEQ, DMODEL, DMODEL);
    vtrans_kernel<<<(SEQ * DMODEL + 255) / 256, 256, 0, stream>>>(vb, vt);
    attn_kernel<<<attn_blocks, 256, 0, stream>>>(qb, kb, vt, ctxb);
    gemm_kernel<false, true, false><<<gD, 256, 0, stream>>>(
        ctxb, wo_bf + l * DD, bo + l * DMODEL, h, h, nullptr,
        SEQ, DMODEL, DMODEL);
    ln_kernel<<<SEQ, 256, 0, stream>>>(h, ln2g + l * DMODEL, ln2b + l * DMODEL, hn);
    gemm_kernel<true, false, true><<<gF, 256, 0, stream>>>(
        hn, w1_bf + l * DF, b1 + l * DFF, nullptr, nullptr, ffb,
        SEQ, DFF, DMODEL);
    gemm_kernel<false, true, false><<<gD, 256, 0, stream>>>(
        ffb, w2_bf + l * DF, b2 + l * DMODEL, h, h, nullptr,
        SEQ, DMODEL, DFF);
  }

  ln_kernel<<<SEQ, 256, 0, stream>>>(h, lnfg, lnfb, hn);
  gemm_kernel<false, false, false><<<gV, 256, 0, stream>>>(
      hn, wh_bf, bhead, nullptr, (float*)d_out, nullptr, SEQ, VOCAB, DMODEL);
}
